// MultiHeadAttention_45543833207190
// MI455X (gfx1250) — matrix operand order repaired and checked by reading, not compiled, not run on hardware
//
#include <hip/hip_runtime.h>

// ---------------- problem constants ----------------
#define S_LEN   2048
#define BATCH   2
#define EMBED   1024
#define HEADS   16
#define HDIM    64
#define NBH     (BATCH*HEADS)            // 32
#define NSD     ((size_t)NBH*S_LEN*HDIM) // 4,194,304 elems per Q/K/V plane
#define SCALE_Q 0.125f                   // 64^-0.5

typedef __attribute__((ext_vector_type(8)))  __bf16 v8bf;
typedef __attribute__((ext_vector_type(16))) __bf16 v16bf;
typedef __attribute__((ext_vector_type(8)))  float  v8f;
typedef __attribute__((ext_vector_type(4)))  unsigned int u32x4;
typedef __attribute__((ext_vector_type(8)))  int i32x8;
typedef __attribute__((ext_vector_type(4)))  int i32x4;
typedef unsigned short u16;

// fp32 -> bf16 via hardware convert (RNE)
__device__ __forceinline__ u16 f2bf(float f) {
    union { __bf16 b; u16 u; } c;
    c.b = (__bf16)f;
    return c.u;
}

// build a 16-element bf16 fragment from two 16-byte LDS chunks
__device__ __forceinline__ v16bf ld_frag(const u16* p0, const u16* p1) {
    union { v16bf v; v8bf h[2]; } u;
    u.h[0] = *(const v8bf*)p0;
    u.h[1] = *(const v8bf*)p1;
    return u.v;
}

__device__ __forceinline__ v8f wmma_bf16(v16bf a, v16bf b, v8f c) {
    return __builtin_amdgcn_wmma_f32_16x16x32_bf16(false, a, false, b,
                                                   (short)0, c, false, false);
}

// ---------------------------------------------------------------------
// Tensor Data Mover: load a (possibly strided-2D) bf16 tile into LDS.
// D# packed per CDNA5 ISA ch.8: group0 {count,lds,global,type}, group1
// {data_size=2B, tensor_dim0/1, tile_dim0/1, dim0 stride}. groups2/3 = 0.
// ---------------------------------------------------------------------
__device__ __forceinline__ void tdm_load(unsigned lds_byte, const void* g,
                                         unsigned td0, unsigned td1,
                                         unsigned tl0, unsigned tl1,
                                         unsigned long long stride0)
{
    unsigned long long ga = (unsigned long long)g;
    u32x4 g0 = { 1u,                                     // count=1, user mode
                 lds_byte,                               // lds_addr [63:32]
                 (unsigned)(ga & 0xffffffffu),           // global_addr lo
                 (unsigned)(((ga >> 32) & 0x1ffffffu)    // global_addr[56:32]
                            | (2u << 30)) };             // type=2 (image)
    i32x8 g1 = { (int)(1u << 16),                        // data_size=1 -> 2B
                 (int)((td0 & 0xffffu) << 16),           // tensor_dim0[15:0]
                 (int)(((td0 >> 16) & 0xffffu) | ((td1 & 0xffffu) << 16)),
                 (int)(((td1 >> 16) & 0xffffu) | ((tl0 & 0xffffu) << 16)),
                 (int)(tl1 & 0xffffu),                   // tile_dim1 (0 = 1D)
                 (int)(stride0 & 0xffffffffu),           // dim0 stride lo
                 (int)((stride0 >> 32) & 0xffffu),       // dim0 stride hi
                 0 };
    i32x4 z4 = {0, 0, 0, 0};
#if __has_include(<hip/amd_detail/amd_gfx1250_TDM.h>)
    i32x8 z8 = {0, 0, 0, 0, 0, 0, 0, 0};
    __builtin_amdgcn_tensor_load_to_lds(g0, g1, z4, z4, z8, 0);
#else
    __builtin_amdgcn_tensor_load_to_lds(g0, g1, z4, z4, 0);
#endif
}

// =====================================================================
// Kernel 1: qkv = x @ W + b ; Q scaled; scattered to bf16 planes:
//   Q,K -> [n][s][d]   V -> [n][d][s] (pre-transposed for PV stage)
//   A = x viewed as [4096,1024], B = W [1024,3072], tile 128x128, K-tile 32
// =====================================================================
__global__ __launch_bounds__(256)
void qkv_gemm(const float* __restrict__ x, const float* __restrict__ W,
              const float* __restrict__ bias, u16* __restrict__ qkv_ws)
{
    __shared__ __align__(16) u16 Asm[128*32];  // row-major [row][k]
    __shared__ __align__(16) u16 Bsm[128*32];  // col-major [col][k]

    const int tid  = threadIdx.x;
    const int lane = tid & 31;
    const int w    = tid >> 5;
    const int wm   = w >> 1;      // 0..3  (row group of 32)
    const int wn   = w & 1;       // 0..1  (col group of 64)
    const bool hi  = (lane & 16) != 0;
    const int m0   = blockIdx.y * 128;
    const int n0   = blockIdx.x * 128;

    v8f acc[2][4];
    #pragma unroll
    for (int i = 0; i < 2; ++i)
        #pragma unroll
        for (int j = 0; j < 4; ++j)
            acc[i][j] = (v8f){0.f,0.f,0.f,0.f,0.f,0.f,0.f,0.f};

    for (int kt = 0; kt < 32; ++kt) {
        // prefetch next K-tile (global_prefetch_b8)
        if (kt + 1 < 32) {
            __builtin_prefetch(x + (size_t)(m0 + (tid >> 1))*EMBED + (kt+1)*32, 0, 0);
            __builtin_prefetch(W + (size_t)((kt+1)*32 + (tid >> 3))*(3*EMBED) + n0, 0, 0);
        }
        // ---- stage A tile: 128x32 fp32 -> bf16 ----
        #pragma unroll
        for (int it = 0; it < 4; ++it) {
            int idx = tid + it*256;            // 0..1023 float4 chunks
            int row = idx >> 3, c4 = (idx & 7) * 4;
            float4 v = *(const float4*)(x + (size_t)(m0+row)*EMBED + kt*32 + c4);
            u16* d = &Asm[row*32 + c4];
            d[0]=f2bf(v.x); d[1]=f2bf(v.y); d[2]=f2bf(v.z); d[3]=f2bf(v.w);
        }
        // ---- stage B tile: 32x128 fp32 -> bf16, transposed to [col][k] ----
        #pragma unroll
        for (int it = 0; it < 4; ++it) {
            int idx = tid + it*256;
            int k = idx >> 5, c4 = (idx & 31) * 4;
            float4 v = *(const float4*)(W + (size_t)(kt*32 + k)*(3*EMBED) + n0 + c4);
            Bsm[(c4+0)*32 + k] = f2bf(v.x);
            Bsm[(c4+1)*32 + k] = f2bf(v.y);
            Bsm[(c4+2)*32 + k] = f2bf(v.z);
            Bsm[(c4+3)*32 + k] = f2bf(v.w);
        }
        __syncthreads();

        v16bf afr[2];
        #pragma unroll
        for (int i = 0; i < 2; ++i) {
            int row  = (wm*2 + i)*16 + (lane & 15);
            int base = row*32 + (hi ? 8 : 0);      // ushort units
            afr[i] = ld_frag(&Asm[base], &Asm[base + 16]);
        }
        #pragma unroll
        for (int j = 0; j < 4; ++j) {
            int col  = (wn*4 + j)*16 + (lane & 15);
            int base = col*32 + (hi ? 8 : 0);
            v16bf bfr = ld_frag(&Bsm[base], &Bsm[base + 16]);
            #pragma unroll
            for (int i = 0; i < 2; ++i)
                acc[i][j] = wmma_bf16(afr[i], bfr, acc[i][j]);
        }
        __syncthreads();
    }

    // ---- epilogue: +bias, scale Q, scatter bf16 ----
    #pragma unroll
    for (int j = 0; j < 4; ++j) {
        int f = n0 + (wn*4 + j)*16 + (lane & 15);     // 0..3071
        float bj   = bias[f];
        int part   = f >> 10;
        int c      = f & 1023;
        int h      = c >> 6;
        int d      = c & 63;
        bool isq   = (part == 0);
        bool isv   = (part == 2);
        u16* base  = qkv_ws + (size_t)part * NSD;
        #pragma unroll
        for (int i = 0; i < 2; ++i) {
            #pragma unroll
            for (int r = 0; r < 8; ++r) {
                int m = m0 + (wm*2 + i)*16 + r + (hi ? 8 : 0);
                float v = acc[i][j][r] + bj;
                if (isq) v *= SCALE_Q;
                int s = m >> 1, b = m & 1;
                int n = b*HEADS + h;
                size_t idx = isv ? ((size_t)n*HDIM + d)*S_LEN + s    // V^T plane
                                 : ((size_t)n*S_LEN + s)*HDIM + d;   // Q/K planes
                base[idx] = f2bf(v);
            }
        }
    }
}

// =====================================================================
// Kernel 2: flash attention, one WG per (head-batch n, 128-row Q block).
// Q/K/V tiles staged into LDS by the Tensor Data Mover.
// =====================================================================
__global__ __launch_bounds__(256)
void flash_attn(const u16* __restrict__ qkv_ws, float* __restrict__ out)
{
    __shared__ __align__(16) u16   Qs[128*64];   // [qrow][d]   16KB
    __shared__ __align__(16) u16   Ks[64*64];    // [kvrow][d]   8KB
    __shared__ __align__(16) u16   Vs[64*64];    // [d][kvrow]   8KB
    __shared__ __align__(16) float Ssc[128*64];  // scores      32KB
    __shared__ __align__(16) u16   Ps[128*64];   // exp probs   16KB
    __shared__ float rowA[128];
    __shared__ float rowL[128];

    const int tid  = threadIdx.x;
    const int lane = tid & 31;
    const int w    = tid >> 5;           // wave = q row tile (0..7)
    const bool hi  = (lane & 16) != 0;
    const int q0   = blockIdx.x * 128;
    const int n    = blockIdx.y;         // head-batch 0..31

    const u16* Qw = qkv_ws;
    const u16* Kw = qkv_ws + NSD;
    const u16* Vw = qkv_ws + 2*NSD;      // [n][d][s] layout

    const unsigned ldsQ = (unsigned)(size_t)(void*)Qs;
    const unsigned ldsK = (unsigned)(size_t)(void*)Ks;
    const unsigned ldsV = (unsigned)(size_t)(void*)Vs;

    // ---- TDM: stage Q block [128][64] bf16 (contiguous 16KB, 1D tile) ----
    if (w == 0) {
        tdm_load(ldsQ, Qw + ((size_t)n*S_LEN + q0)*HDIM,
                 /*td0*/128*HDIM, /*td1*/1, /*tl0*/128*HDIM, /*tl1*/0,
                 /*stride0*/128*HDIM);
    }

    float m_i = -3.0e38f, l_i = 0.0f;    // valid for tid < 128
    v8f o[4];
    #pragma unroll
    for (int j = 0; j < 4; ++j) o[j] = (v8f){0.f,0.f,0.f,0.f,0.f,0.f,0.f,0.f};

    for (int kb = 0; kb < 32; ++kb) {
        const int kv0 = kb * 64;
        // ---- TDM staging: wave0 -> K (1D contiguous), wave1 -> V (2D strided)
        if (w == 0) {
            tdm_load(ldsK, Kw + ((size_t)n*S_LEN + kv0)*HDIM,
                     /*td0*/64*HDIM, /*td1*/1, /*tl0*/64*HDIM, /*tl1*/0,
                     /*stride0*/64*HDIM);
            __builtin_amdgcn_s_wait_tensorcnt(0);
        } else if (w == 1) {
            tdm_load(ldsV, Vw + (size_t)n*HDIM*S_LEN + kv0,
                     /*td0*/S_LEN, /*td1*/HDIM, /*tl0*/64, /*tl1*/64,
                     /*stride0*/S_LEN);
            __builtin_amdgcn_s_wait_tensorcnt(0);
        }
        __syncthreads();

        // ---- S = Q Kt : wave w -> rows [w*16, w*16+16), all 64 kv cols ----
        v8f sacc[4];
        #pragma unroll
        for (int nt = 0; nt < 4; ++nt) sacc[nt] = (v8f){0.f,0.f,0.f,0.f,0.f,0.f,0.f,0.f};
        #pragma unroll
        for (int ks = 0; ks < 2; ++ks) {
            int arow = w*16 + (lane & 15);
            int ab   = arow*64 + ks*32 + (hi ? 8 : 0);
            v16bf af = ld_frag(&Qs[ab], &Qs[ab + 16]);
            #pragma unroll
            for (int nt = 0; nt < 4; ++nt) {
                int col = nt*16 + (lane & 15);
                int bb  = col*64 + ks*32 + (hi ? 8 : 0);
                v16bf bf = ld_frag(&Ks[bb], &Ks[bb + 16]);
                sacc[nt] = wmma_bf16(af, bf, sacc[nt]);
            }
        }
        #pragma unroll
        for (int nt = 0; nt < 4; ++nt)
            #pragma unroll
            for (int r = 0; r < 8; ++r)
                Ssc[(w*16 + r + (hi ? 8 : 0))*64 + nt*16 + (lane & 15)] = sacc[nt][r];
        __syncthreads();

        // ---- online softmax: one thread per q row ----
        if (tid < 128) {
            const float* sr = &Ssc[tid*64];
            float mx = m_i;
            for (int jx = 0; jx < 64; ++jx) mx = fmaxf(mx, sr[jx]);
            float alpha = __expf(m_i - mx);
            float l = l_i * alpha;
            for (int jx = 0; jx < 64; ++jx) {
                float p = __expf(sr[jx] - mx);
                l += p;
                Ps[tid*64 + jx] = f2bf(p);
            }
            m_i = mx; l_i = l;
            rowA[tid] = alpha;
        }
        __syncthreads();

        // ---- rescale running output, then O += P @ V ----
        float ar[8];
        #pragma unroll
        for (int r = 0; r < 8; ++r) ar[r] = rowA[w*16 + r + (hi ? 8 : 0)];
        #pragma unroll
        for (int j = 0; j < 4; ++j)
            #pragma unroll
            for (int r = 0; r < 8; ++r)
                o[j][r] *= ar[r];

        #pragma unroll
        for (int ks = 0; ks < 2; ++ks) {
            int arow = w*16 + (lane & 15);
            int ab   = arow*64 + ks*32 + (hi ? 8 : 0);
            v16bf af = ld_frag(&Ps[ab], &Ps[ab + 16]);
            #pragma unroll
            for (int j = 0; j < 4; ++j) {
                int col = j*16 + (lane & 15);           // d tile
                int bb  = col*64 + ks*32 + (hi ? 8 : 0);
                v16bf bf = ld_frag(&Vs[bb], &Vs[bb + 16]);
                o[j] = wmma_bf16(af, bf, o[j]);
            }
        }
        __syncthreads();
    }

    if (tid < 128) rowL[tid] = l_i;
    __syncthreads();

    float lr[8];
    #pragma unroll
    for (int r = 0; r < 8; ++r) lr[r] = 1.0f / rowL[w*16 + r + (hi ? 8 : 0)];

    const int b = n >> 4, h = n & 15;
    #pragma unroll
    for (int j = 0; j < 4; ++j) {
        int d = j*16 + (lane & 15);
        #pragma unroll
        for (int r = 0; r < 8; ++r) {
            int srow = q0 + w*16 + r + (hi ? 8 : 0);
            out[((size_t)srow*BATCH + b)*EMBED + h*HDIM + d] = o[j][r] * lr[r];
        }
    }
}

// =====================================================================
extern "C" void kernel_launch(void* const* d_in, const int* in_sizes, int n_in,
                              void* d_out, int out_size, void* d_ws, size_t ws_size,
                              hipStream_t stream)
{
    const float* x    = (const float*)d_in[0];   // [2048, 2, 1024]
    const float* W    = (const float*)d_in[1];   // [1024, 3072]
    const float* bias = (const float*)d_in[2];   // [3072]
    float*       out  = (float*)d_out;           // [2048, 2, 1024]
    u16*         qkv  = (u16*)d_ws;              // 3 * NSD bf16 = 24 MB

    dim3 g1(3*EMBED/128, (S_LEN*BATCH)/128);     // 24 x 32
    qkv_gemm<<<g1, dim3(256), 0, stream>>>(x, W, bias, qkv);

    dim3 g2(S_LEN/128, NBH);                     // 16 x 32
    flash_attn<<<g2, dim3(256), 0, stream>>>(qkv, out);
}
